// MetabolicBrain_64613488001032
// MI455X (gfx1250) — matrix operand order repaired and checked by reading, not compiled, not run on hardware
//
#include <hip/hip_runtime.h>
#include <math.h>

#define BB   32
#define TT   2048
#define DD   128
#define VV   256
#define MMEM 256
#define NTHR 512

typedef __attribute__((ext_vector_type(16))) __bf16 v16bf;
typedef __attribute__((ext_vector_type(8)))  float  v8f;

// ---------------- d_out offsets (in floats) ----------------
#define LOGITS_OFF 0            // [32,2048,256]
#define HF_OFF     16777216     // [1,32,128]
#define HM_OFF     16781312     // [32,128]
#define DM_OFF     16785408     // depths.mean()
#define CS_OFF     16785409     // costs.sum()
#define VM_OFF     16785410     // volms[-1]
#define COMB_OFF   16785411     // [32,2048,128]
#define SAL_OFF    25174019     // [32,2048,1]

// ---------------- d_ws offsets (bytes) ----------------
#define WS_GWIH  0              // gru_w_ih bf16 [384][128]
#define WS_GWHH  98304
#define WS_WWIH  196608
#define WS_WWHH  294912
#define WS_VGW1  393216         // [64][128]
#define WS_THW1  409600         // [16][128]
#define WS_KEYS  413696         // keys_n bf16 [256][128]
#define WS_MVT   479232         // mem_vals^T bf16 [128][256]
#define WS_SOMA  544768         // soma_w bf16 [256][128]

// ---------------- LDS layout (bytes) ----------------
#define SM_WWIH  0
#define SM_WWHH  98304
#define SM_E     196608     // aliased as REL after depth loop
#define SM_HF    212992
#define SM_HM    229376
#define SM_LAT   245760
#define SM_ATT   262144
#define SM_HID   294912
#define SM_FEAT  303104
#define SM_BIHG  305152
#define SM_BHHG  306688
#define SM_BIHW  308224
#define SM_BHHW  309760
#define SM_VGB1  311296
#define SM_VGW2  311552
#define SM_ONG   311808
#define SM_ONB   312320
#define SM_HARD  312832
#define SM_THB1  313856
#define SM_THW2  313920
#define SM_VOLN  313984
#define SM_INVQ  314112
#define SM_RS    314240
#define SM_SCAL  314368
#define SMEM_BYTES 314432

// scal slots: 0 stopped, 1 curMaxD, 2 meanVol, 3 lastD, 4 sumDepth, 5 sumCost, 6 sBias, 7 gateActive

// ================= helpers =================
__device__ __forceinline__ v8f zero8() {
  v8f r;
#pragma unroll
  for (int i = 0; i < 8; i++) r[i] = 0.0f;
  return r;
}
// A fragment (16x32 bf16) from fp32 row-major source; lane holds one row,
// K split: chunk [half..half+7] and [16+half..16+half+7]  (half = 8 for hi lanes)
// Both 8-float chunks are 32B aligned -> 4x b128 loads + pack converts.
__device__ __forceinline__ v16bf fragA_f32(const float* p) {
  const float4* q = (const float4*)p;
  float4 c0 = q[0], c1 = q[1], c2 = q[4], c3 = q[5];
  v16bf r;
  r[0] = (__bf16)c0.x;  r[1] = (__bf16)c0.y;  r[2]  = (__bf16)c0.z;  r[3]  = (__bf16)c0.w;
  r[4] = (__bf16)c1.x;  r[5] = (__bf16)c1.y;  r[6]  = (__bf16)c1.z;  r[7]  = (__bf16)c1.w;
  r[8] = (__bf16)c2.x;  r[9] = (__bf16)c2.y;  r[10] = (__bf16)c2.z;  r[11] = (__bf16)c2.w;
  r[12] = (__bf16)c3.x; r[13] = (__bf16)c3.y; r[14] = (__bf16)c3.z;  r[15] = (__bf16)c3.w;
  return r;
}
__device__ __forceinline__ v16bf fragA_f32s(const float* p, float s) {
  const float4* q = (const float4*)p;
  float4 c0 = q[0], c1 = q[1], c2 = q[4], c3 = q[5];
  v16bf r;
  r[0] = (__bf16)(c0.x * s);  r[1] = (__bf16)(c0.y * s);
  r[2] = (__bf16)(c0.z * s);  r[3] = (__bf16)(c0.w * s);
  r[4] = (__bf16)(c1.x * s);  r[5] = (__bf16)(c1.y * s);
  r[6] = (__bf16)(c1.z * s);  r[7] = (__bf16)(c1.w * s);
  r[8] = (__bf16)(c2.x * s);  r[9] = (__bf16)(c2.y * s);
  r[10] = (__bf16)(c2.z * s); r[11] = (__bf16)(c2.w * s);
  r[12] = (__bf16)(c3.x * s); r[13] = (__bf16)(c3.y * s);
  r[14] = (__bf16)(c3.z * s); r[15] = (__bf16)(c3.w * s);
  return r;
}
// B fragment (32x16 bf16): lane holds one column, K split like A:
// chunk [half..half+7] and [16+half..16+half+7] -> 2x b128 loads (16B aligned).
__device__ __forceinline__ v16bf fragB_bf(const __bf16* p) {
  typedef __attribute__((ext_vector_type(8))) __bf16 v8bf_t;
  v8bf_t lo = *(const v8bf_t*)p;
  v8bf_t hi = *(const v8bf_t*)(p + 16);
  return __builtin_shufflevector(lo, hi, 0, 1, 2, 3, 4, 5, 6, 7, 8, 9, 10, 11, 12, 13, 14, 15);
}
__device__ __forceinline__ v8f wmma_bf16(v16bf a, v16bf b, v8f c) {
  return __builtin_amdgcn_wmma_f32_16x16x32_bf16(false, a, false, b, (short)0, c, false, false);
}
__device__ __forceinline__ float sigm(float x) { return 1.0f / (1.0f + expf(-x)); }
__device__ __forceinline__ float gelu_exact(float x) {
  return 0.5f * x * (1.0f + erff(x * 0.70710678118654752f));
}
__device__ __forceinline__ float hsum16(float v) {  // sum within 16-lane half-wave
  v += __shfl_xor(v, 1); v += __shfl_xor(v, 2); v += __shfl_xor(v, 4); v += __shfl_xor(v, 8);
  return v;
}
__device__ __forceinline__ float hmax16(float v) {
  v = fmaxf(v, __shfl_xor(v, 1)); v = fmaxf(v, __shfl_xor(v, 2));
  v = fmaxf(v, __shfl_xor(v, 4)); v = fmaxf(v, __shfl_xor(v, 8));
  return v;
}

// One full GRU step: Hl <- GRU(Xl, Hl).  All 16 waves participate.
// wave -> (m = wv>>3, nt = wv&7): owns rows m*16..+15, cols nt*16..+15 of D=128.
// kb loop kept rolled (#pragma unroll 1) to bound register pressure: at any time
// 6 accumulators + 2 A-frags + 6 B-frags ~ 120 VGPRs -> no scratch spills.
__device__ __forceinline__ void gru_step(
    const float* __restrict__ Xl, float* __restrict__ Hl,
    const __bf16* __restrict__ Wih, const __bf16* __restrict__ Whh,
    const float* __restrict__ bih, const float* __restrict__ bhh,
    int lane, int wv) {
  const int m = wv >> 3, nt = wv & 7;
  const int li = lane & 15;
  const int halfA = (lane >> 4) << 3;
  const int halfB = (lane >> 4) << 4;
  const int rowA = (m << 4) + li;
  const int colD = (nt << 4) + li;
  v8f aRi = zero8(), aRh = zero8(), aZi = zero8(), aZh = zero8(), aNi = zero8(), aNh = zero8();
#pragma unroll 1
  for (int kb = 0; kb < 4; kb++) {
    const float* pax = Xl + rowA * DD + kb * 32 + halfA;
    const float* pah = Hl + rowA * DD + kb * 32 + halfA;
    const __bf16* wb = Wih + colD * DD + kb * 32 + halfA;
    const __bf16* ub = Whh + colD * DD + kb * 32 + halfA;
    v16bf ax = fragA_f32(pax);
    v16bf ah = fragA_f32(pah);
    v16bf br = fragB_bf(wb);
    v16bf bz = fragB_bf(wb + 128 * DD);
    v16bf bn = fragB_bf(wb + 256 * DD);
    v16bf ur = fragB_bf(ub);
    v16bf uz = fragB_bf(ub + 128 * DD);
    v16bf un = fragB_bf(ub + 256 * DD);
    aRi = wmma_bf16(ax, br, aRi);
    aZi = wmma_bf16(ax, bz, aZi);
    aNi = wmma_bf16(ax, bn, aNi);
    aRh = wmma_bf16(ah, ur, aRh);
    aZh = wmma_bf16(ah, uz, aZh);
    aNh = wmma_bf16(ah, un, aNh);
  }
  const float bir = bih[colD], bhr = bhh[colD];
  const float biz = bih[128 + colD], bhz = bhh[128 + colD];
  const float bin_ = bih[256 + colD], bhn = bhh[256 + colD];
  float hp[8];
#pragma unroll
  for (int v = 0; v < 8; v++) {
    int row = (m << 4) + v + halfA;
    float r = sigm(aRi[v] + bir + aRh[v] + bhr);
    float z = sigm(aZi[v] + biz + aZh[v] + bhz);
    float n = tanhf(aNi[v] + bin_ + r * (aNh[v] + bhn));
    hp[v] = (1.0f - z) * n + z * Hl[row * DD + colD];
  }
  __syncthreads();
#pragma unroll
  for (int v = 0; v < 8; v++) {
    int row = (m << 4) + v + halfA;
    Hl[row * DD + colD] = hp[v];
  }
  __syncthreads();
}

// ================= prep kernels =================
__global__ void cvt_bf16(const float* __restrict__ s, __bf16* __restrict__ d, int n) {
  int i = blockIdx.x * blockDim.x + threadIdx.x;
  if (i < n) d[i] = (__bf16)s[i];
}
__global__ void keysn_kernel(const float* __restrict__ keys, __bf16* __restrict__ d) {
  int j = blockIdx.x, k = threadIdx.x;  // 256 blocks x 128 threads
  __shared__ float red[4];
  float v = keys[j * DD + k];
  float s = v * v;
#pragma unroll
  for (int m = 1; m < 32; m <<= 1) s += __shfl_xor(s, m);
  if ((k & 31) == 0) red[k >> 5] = s;
  __syncthreads();
  float tot = red[0] + red[1] + red[2] + red[3];
  d[j * DD + k] = (__bf16)(v / fmaxf(sqrtf(tot), 1e-12f));
}
__global__ void mvT_kernel(const float* __restrict__ mv, __bf16* __restrict__ d) {
  int i = blockIdx.x * blockDim.x + threadIdx.x;  // over 128*256
  if (i < DD * MMEM) {
    int n = i >> 8, k = i & 255;
    d[n * MMEM + k] = (__bf16)mv[(size_t)k * DD + n];
  }
}

// ================= persistent recurrence kernel (1 workgroup) =================
__global__ __launch_bounds__(NTHR) void brain_kernel(
    const int* __restrict__ xin, const float* __restrict__ hf_in,
    const float* __restrict__ hm_in, const float* __restrict__ surprise,
    const float* __restrict__ soma_w,
    const float* __restrict__ gbih, const float* __restrict__ gbhh,
    const float* __restrict__ wbih, const float* __restrict__ wbhh,
    const float* __restrict__ vgb1_g, const float* __restrict__ vgw2_g,
    const float* __restrict__ vgb2_g,
    const float* __restrict__ ong_g, const float* __restrict__ onb_g,
    const float* __restrict__ thb1_g, const float* __restrict__ thw2_g,
    const float* __restrict__ thb2_g,
    const float* __restrict__ hard_g,
    const __bf16* __restrict__ gWih, const __bf16* __restrict__ gWhh,
    const __bf16* __restrict__ wWih_g, const __bf16* __restrict__ wWhh_g,
    const __bf16* __restrict__ vgW1, const __bf16* __restrict__ thW1,
    const __bf16* __restrict__ keysn, const __bf16* __restrict__ mvT,
    float* __restrict__ dout) {
  extern __shared__ char smem[];
  __bf16* wWih = (__bf16*)(smem + SM_WWIH);
  __bf16* wWhh = (__bf16*)(smem + SM_WWHH);
  float* E    = (float*)(smem + SM_E);
  float* REL  = E;  // alias: rel only lives after E's last use in a step
  float* Hf   = (float*)(smem + SM_HF);
  float* Hm   = (float*)(smem + SM_HM);
  float* Lat  = (float*)(smem + SM_LAT);
  float* ATT  = (float*)(smem + SM_ATT);
  float* HID  = (float*)(smem + SM_HID);
  float* FEAT = (float*)(smem + SM_FEAT);
  float* bihG = (float*)(smem + SM_BIHG);
  float* bhhG = (float*)(smem + SM_BHHG);
  float* bihW = (float*)(smem + SM_BIHW);
  float* bhhW = (float*)(smem + SM_BHHW);
  float* vgB1 = (float*)(smem + SM_VGB1);
  float* vgW2l = (float*)(smem + SM_VGW2);
  float* onG  = (float*)(smem + SM_ONG);
  float* onB  = (float*)(smem + SM_ONB);
  float* hard10 = (float*)(smem + SM_HARD);
  float* thB1 = (float*)(smem + SM_THB1);
  float* thW2l = (float*)(smem + SM_THW2);
  float* voln = (float*)(smem + SM_VOLN);
  float* invq = (float*)(smem + SM_INVQ);
  float* rs   = (float*)(smem + SM_RS);
  float* scal = (float*)(smem + SM_SCAL);

  const int tid = threadIdx.x, lane = tid & 31, wv = tid >> 5;

  {  // stage ws weights (hot: used up to 5x per step) into LDS, 128b copies
    const uint4* s = (const uint4*)wWih_g; uint4* d = (uint4*)wWih;
    for (int i = tid; i < 6144; i += NTHR) d[i] = s[i];
    s = (const uint4*)wWhh_g; d = (uint4*)wWhh;
    for (int i = tid; i < 6144; i += NTHR) d[i] = s[i];
  }
  for (int i = tid; i < BB * DD; i += NTHR) { Hf[i] = hf_in[i]; Hm[i] = hm_in[i]; }
  for (int i = tid; i < 384; i += NTHR) {
    bihG[i] = gbih[i]; bhhG[i] = gbhh[i]; bihW[i] = wbih[i]; bhhW[i] = wbhh[i];
  }
  for (int i = tid; i < DD; i += NTHR) { onG[i] = ong_g[i]; onB[i] = onb_g[i]; }
  for (int i = tid; i < MMEM; i += NTHR) hard10[i] = hard_g[i] * 10.0f;
  for (int i = tid; i < 64; i += NTHR) { vgB1[i] = vgb1_g[i]; vgW2l[i] = vgw2_g[i]; }
  if (tid < 16) { thB1[tid] = thb1_g[tid]; thW2l[tid] = thw2_g[tid]; }
  if (tid == 0) {
    float sp = surprise[0];
    scal[6] = logf(fmaxf(1.0f, sp));         // s_bias (DOPA=1)
    scal[7] = 0.4f + 0.2f * sigm(sp);        // gate_active
    scal[4] = 0.0f; scal[5] = 0.0f;          // sumDepth, sumCost
  }
  __syncthreads();

  const int li = lane & 15;
  const int halfA = (lane >> 4) << 3;
  const int halfB = (lane >> 4) << 4;

  for (int t = 0; t < TT; t++) {
    // ---- embedding gather: E[b][d] = soma_w[x[b,t]][d]
    for (int i = tid; i < BB * DD; i += NTHR) {
      int b = i >> 7, d = i & 127;
      E[i] = soma_w[xin[b * TT + t] * DD + d];
    }
    __syncthreads();

    // ---- forward GRU: Hf <- GRU(E, Hf); Hf is now p_t
    gru_step(E, Hf, gWih, gWhh, bihG, bhhG, lane, wv);

    // ---- thalamus: feat = gelu(p_t @ th_w1^T + b1)  (waves 0,1)
    if (wv < 2) {
      int m = wv;
      int rowA = (m << 4) + li;
      v8f acc = zero8();
#pragma unroll 1
      for (int kb = 0; kb < 4; kb++) {
        v16bf a = fragA_f32(Hf + rowA * DD + kb * 32 + halfA);
        acc = wmma_bf16(a, fragB_bf(thW1 + li * DD + kb * 32 + halfA), acc);
      }
      float b1 = thB1[li];
#pragma unroll
      for (int v = 0; v < 8; v++) {
        int row = (m << 4) + v + halfA;
        FEAT[row * 16 + li] = gelu_exact(acc[v] + b1);
      }
    }
    __syncthreads();
    if (tid < BB) {
      float s = thb2_g[0];
#pragma unroll
      for (int k = 0; k < 16; k++) s += FEAT[tid * 16 + k] * thW2l[k];
      rs[tid] = sigm(s);
    }
    __syncthreads();
    if (tid == 0) {
      float s = 0.0f;
      for (int b = 0; b < BB; b++) s += rs[b];
      float salv = s * (1.0f / BB);
      scal[1] = fminf(fmaxf(floorf(2.0f + 3.0f * sqrtf(salv)), 2.0f), 5.0f);  // cur_max_d
      scal[0] = 0.0f; scal[2] = 0.0f; scal[3] = 0.0f;  // stopped, meanVol, lastD
    }
    for (int i = tid; i < BB * DD; i += NTHR) Lat[i] = Hf[i];  // latent = p_t
    __syncthreads();

    // ---- depth loop (control scalars are batch-uniform -> uniform branches)
    for (int dd = 0; dd < 5; dd++) {
      bool active = (scal[0] == 0.0f) && ((float)dd < scal[1]);
      if (active) {
        // hid = gelu(Lat @ vg_w1^T + b1)  (waves 0..7)
        if (wv < 8) {
          int m = wv >> 2, nt = wv & 3;
          int rowA = (m << 4) + li, col = (nt << 4) + li;
          v8f acc = zero8();
#pragma unroll 1
          for (int kb = 0; kb < 4; kb++) {
            v16bf a = fragA_f32(Lat + rowA * DD + kb * 32 + halfA);
            acc = wmma_bf16(a, fragB_bf(vgW1 + col * DD + kb * 32 + halfA), acc);
          }
          float b1 = vgB1[col];
#pragma unroll
          for (int v = 0; v < 8; v++) {
            int row = (m << 4) + v + halfA;
            HID[row * 64 + col] = gelu_exact(acc[v] + b1);
          }
        }
        __syncthreads();
        if (tid < BB) {  // vol_new = sigmoid(hid @ vg_w2 + b2 + s_bias)
          float s = vgb2_g[0] + scal[6];
#pragma unroll
          for (int k = 0; k < 64; k++) s += HID[tid * 64 + k] * vgW2l[k];
          voln[tid] = sigm(s);
        }
        // ws GRU: Hm <- GRU(Lat, Hm)  (weights resident in LDS)
        gru_step(Lat, Hm, wWih, wWhh, bihW, bhhW, lane, wv);
        if (tid == 0) {
          float s = 0.0f;
          for (int b = 0; b < BB; b++) s += voln[b];
          float mvn = s * (1.0f / BB);
          scal[2] = mvn; scal[3] = (float)dd;
          if (dd >= 2 && mvn < 0.15f) scal[0] = 1.0f;  // VOL_THRESH
        }
        {  // latent = LN(latent*keep + e*amix + h_new*vol*0.1); half-wave per row
          int row = (wv << 1) + (lane >> 4);
          float amix = (dd == 0) ? 0.4f : 0.15f;
          float keep = 1.0f - amix - 0.1f;
          float vr = voln[row] * 0.1f;
          float vals[8], s = 0.0f;
#pragma unroll
          for (int i = 0; i < 8; i++) {
            int d = li * 8 + i;
            vals[i] = Lat[row * DD + d] * keep + E[row * DD + d] * amix + Hm[row * DD + d] * vr;
            s += vals[i];
          }
          float mu = hsum16(s) * (1.0f / DD);
          float vv = 0.0f;
#pragma unroll
          for (int i = 0; i < 8; i++) { float q = vals[i] - mu; vv += q * q; }
          vv = hsum16(vv) * (1.0f / DD);
          float inv = 1.0f / sqrtf(vv + 1e-5f);
#pragma unroll
          for (int i = 0; i < 8; i++) {
            int d = li * 8 + i;
            Lat[row * DD + d] = (vals[i] - mu) * inv * onG[d] + onB[d];
          }
        }
        __syncthreads();
      }
    }

    if (tid == 0) {  // per-step stats
      float depth = scal[3] + 1.0f;
      scal[4] += depth;
      scal[5] += scal[2] * 0.001f * depth;
    }
    {  // q row inverse norms
      int row = (wv << 1) + (lane >> 4);
      float s = 0.0f;
#pragma unroll
      for (int i = 0; i < 8; i++) { float q = Lat[row * DD + li * 8 + i]; s += q * q; }
      s = hsum16(s);
      if (li == 0) invq[row] = 1.0f / fmaxf(sqrtf(s), 1e-12f);
    }
    __syncthreads();
    // ---- scores = (q @ keys_n^T) * hardness * 10  (2 tiles per wave)
#pragma unroll 1
    for (int rep = 0; rep < 2; rep++) {
      int tc = wv + rep * 16;
      int m = tc & 1, nt = tc >> 1;
      int rowA = (m << 4) + li, n = (nt << 4) + li;
      float qs = invq[rowA];
      v8f acc = zero8();
#pragma unroll 1
      for (int kb = 0; kb < 4; kb++) {
        v16bf a = fragA_f32s(Lat + rowA * DD + kb * 32 + halfA, qs);
        acc = wmma_bf16(a, fragB_bf(keysn + n * DD + kb * 32 + halfA), acc);
      }
      float hs = hard10[n];
#pragma unroll
      for (int v = 0; v < 8; v++) {
        int row = (m << 4) + v + halfA;
        ATT[row * MMEM + n] = acc[v] * hs;
      }
    }
    __syncthreads();
    {  // softmax over 256 per row (half-wave: 16 lanes x 16 elems)
      int row = (wv << 1) + (lane >> 4);
      float v[16], mx = -3.4e38f;
#pragma unroll
      for (int i = 0; i < 16; i++) { v[i] = ATT[row * MMEM + li * 16 + i]; mx = fmaxf(mx, v[i]); }
      mx = hmax16(mx);
      float s = 0.0f;
#pragma unroll
      for (int i = 0; i < 16; i++) { v[i] = expf(v[i] - mx); s += v[i]; }
      s = hsum16(s);
      float inv = 1.0f / s;
#pragma unroll
      for (int i = 0; i < 16; i++) ATT[row * MMEM + li * 16 + i] = v[i] * inv;
    }
    __syncthreads();
    {  // rel = attn @ mem_vals  (1 tile per wave, K=256), write into REL(=E alias)
      int m = wv & 1, nt = wv >> 1;
      int rowA = (m << 4) + li, n = (nt << 4) + li;
      v8f acc = zero8();
#pragma unroll 1
      for (int kb = 0; kb < 8; kb++) {
        v16bf a = fragA_f32(ATT + rowA * MMEM + kb * 32 + halfA);
        acc = wmma_bf16(a, fragB_bf(mvT + n * MMEM + kb * 32 + halfA), acc);
      }
#pragma unroll
      for (int v = 0; v < 8; v++) {
        int row = (m << 4) + v + halfA;
        REL[row * DD + n] = acc[v];
      }
    }
    __syncthreads();
    {  // out = LN(latent + rel*gate_active) -> combined[b][t][:]
      int row = (wv << 1) + (lane >> 4);
      float ga = scal[7];
      float vals[8], s = 0.0f;
#pragma unroll
      for (int i = 0; i < 8; i++) {
        int d = li * 8 + i;
        vals[i] = Lat[row * DD + d] + REL[row * DD + d] * ga;
        s += vals[i];
      }
      float mu = hsum16(s) * (1.0f / DD);
      float vv = 0.0f;
#pragma unroll
      for (int i = 0; i < 8; i++) { float q = vals[i] - mu; vv += q * q; }
      vv = hsum16(vv) * (1.0f / DD);
      float inv = 1.0f / sqrtf(vv + 1e-5f);
      float* outc = dout + COMB_OFF + ((size_t)row * TT + t) * DD;
#pragma unroll
      for (int i = 0; i < 8; i++) {
        int d = li * 8 + i;
        outc[d] = (vals[i] - mu) * inv * onG[d] + onB[d];
      }
    }
    __syncthreads();  // protects REL(=E) before next step's gather
  }

  for (int i = tid; i < BB * DD; i += NTHR) {
    dout[HF_OFF + i] = Hf[i];
    dout[HM_OFF + i] = Hm[i];
  }
  if (tid == 0) {
    dout[DM_OFF] = scal[4] * (1.0f / TT);
    dout[CS_OFF] = scal[5];
    dout[VM_OFF] = scal[2];
  }
}

// ================= parallel logits kernel: logits = combined @ soma_w^T =================
__global__ __launch_bounds__(NTHR) void logits_kernel(
    const float* __restrict__ comb, const __bf16* __restrict__ somab,
    float* __restrict__ logits, float* __restrict__ sal,
    const float* __restrict__ surprise) {
  int t = blockIdx.x;
  int tid = threadIdx.x, lane = tid & 31, wv = tid >> 5;
  if (tid < BB) {
    float ga = 0.4f + 0.2f / (1.0f + expf(-surprise[0]));
    sal[(size_t)tid * TT + t] = ga;
  }
  const int li = lane & 15;
  const int halfA = (lane >> 4) << 3;
  const int halfB = (lane >> 4) << 4;
#pragma unroll 1
  for (int rep = 0; rep < 2; rep++) {
    int tc = wv + rep * 16;
    int m = tc & 1, nt = tc >> 1;
    int rowA = (m << 4) + li, n = (nt << 4) + li;
    const float* arow = comb + ((size_t)rowA * TT + t) * DD;
    v8f acc = zero8();
#pragma unroll 1
    for (int kb = 0; kb < 4; kb++) {
      v16bf a = fragA_f32(arow + kb * 32 + halfA);
      acc = wmma_bf16(a, fragB_bf(somab + n * DD + kb * 32 + halfA), acc);
    }
#pragma unroll
    for (int v = 0; v < 8; v++) {
      int row = (m << 4) + v + halfA;
      logits[((size_t)row * TT + t) * VV + n] = acc[v];
    }
  }
}

// ================= host entry =================
extern "C" void kernel_launch(void* const* d_in, const int* in_sizes, int n_in,
                              void* d_out, int out_size, void* d_ws, size_t ws_size,
                              hipStream_t stream) {
  (void)in_sizes; (void)n_in; (void)out_size; (void)ws_size;
  const int*   x     = (const int*)d_in[0];
  const float* hf    = (const float*)d_in[1];
  const float* hm    = (const float*)d_in[2];
  const float* sur   = (const float*)d_in[3];
  const float* soma  = (const float*)d_in[4];
  const float* gwih  = (const float*)d_in[5];
  const float* gwhh  = (const float*)d_in[6];
  const float* gbih  = (const float*)d_in[7];
  const float* gbhh  = (const float*)d_in[8];
  const float* wwih  = (const float*)d_in[9];
  const float* wwhh  = (const float*)d_in[10];
  const float* wbih  = (const float*)d_in[11];
  const float* wbhh  = (const float*)d_in[12];
  const float* vgw1  = (const float*)d_in[13];
  const float* vgb1  = (const float*)d_in[14];
  const float* vgw2  = (const float*)d_in[15];
  const float* vgb2  = (const float*)d_in[16];
  const float* ong   = (const float*)d_in[17];
  const float* onb   = (const float*)d_in[18];
  const float* thw1  = (const float*)d_in[19];
  const float* thb1  = (const float*)d_in[20];
  const float* thw2  = (const float*)d_in[21];
  const float* thb2  = (const float*)d_in[22];
  const float* mkeys = (const float*)d_in[23];
  const float* mvals = (const float*)d_in[24];
  const float* mhard = (const float*)d_in[25];

  char* ws = (char*)d_ws;
  __bf16* gWihB = (__bf16*)(ws + WS_GWIH);
  __bf16* gWhhB = (__bf16*)(ws + WS_GWHH);
  __bf16* wWihB = (__bf16*)(ws + WS_WWIH);
  __bf16* wWhhB = (__bf16*)(ws + WS_WWHH);
  __bf16* vgW1B = (__bf16*)(ws + WS_VGW1);
  __bf16* thW1B = (__bf16*)(ws + WS_THW1);
  __bf16* keysB = (__bf16*)(ws + WS_KEYS);
  __bf16* mvTB  = (__bf16*)(ws + WS_MVT);
  __bf16* somaB = (__bf16*)(ws + WS_SOMA);
  float* out = (float*)d_out;

  cvt_bf16<<<(49152 + 255) / 256, 256, 0, stream>>>(gwih, gWihB, 49152);
  cvt_bf16<<<(49152 + 255) / 256, 256, 0, stream>>>(gwhh, gWhhB, 49152);
  cvt_bf16<<<(49152 + 255) / 256, 256, 0, stream>>>(wwih, wWihB, 49152);
  cvt_bf16<<<(49152 + 255) / 256, 256, 0, stream>>>(wwhh, wWhhB, 49152);
  cvt_bf16<<<(8192 + 255) / 256, 256, 0, stream>>>(vgw1, vgW1B, 8192);
  cvt_bf16<<<(2048 + 255) / 256, 256, 0, stream>>>(thw1, thW1B, 2048);
  cvt_bf16<<<(32768 + 255) / 256, 256, 0, stream>>>(soma, somaB, 32768);
  keysn_kernel<<<256, 128, 0, stream>>>(mkeys, keysB);
  mvT_kernel<<<(32768 + 255) / 256, 256, 0, stream>>>(mvals, mvTB);

  (void)hipFuncSetAttribute((const void*)brain_kernel,
                            hipFuncAttributeMaxDynamicSharedMemorySize, SMEM_BYTES);
  brain_kernel<<<1, NTHR, SMEM_BYTES, stream>>>(
      x, hf, hm, sur, soma, gbih, gbhh, wbih, wbhh, vgb1, vgw2, vgb2, ong, onb,
      thb1, thw2, thb2, mhard, gWihB, gWhhB, wWihB, wWhhB, vgW1B, thW1B, keysB, mvTB, out);

  logits_kernel<<<TT, NTHR, 0, stream>>>(out + COMB_OFF, somaB,
                                         out + LOGITS_OFF, out + SAL_OFF, sur);
}
